// SpatialBlock_23441931501721
// MI455X (gfx1250) — matrix operand order repaired and checked by reading, not compiled, not run on hardware
//
#include <hip/hip_runtime.h>
#include <hip/hip_bf16.h>

typedef __attribute__((ext_vector_type(16))) _Float16 v16h;
typedef __attribute__((ext_vector_type(8)))  float    v8f;

#define DEV __device__ __forceinline__

// Problem constants
constexpr int BN = 4, SN = 12, BS = 48;      // batch, seq, batch*seq
constexpr int N  = 716;                      // nodes
constexpr int NP = 736;                      // padded nodes (multiple of 32)
constexpr int D  = 64, H = 4, DH = 16;
constexpr int NT = 45;                       // 16-row tiles (45*16 = 720)
constexpr float NEGV = -9.0e15f;
constexpr float ALPHA = 0.2f;

// --- WMMA fragment index maps (CDNA5 ISA 7.12.2, wave32) ---
// A (16x32 f16): lanes0-15 hold K 0-7 (v0-3) & 16-23 (v4-7); lanes16-31 hold K 8-15 & 24-31. M = lane&15.
DEV int a_koff(int idx, int hi) {
  int v = idx >> 1, e = idx & 1;
  return ((v & 3) * 2) + (hi ? 8 : 0) + ((v >> 2) * 16) + e;
}
// B (32x16 f16): lanes0-15 hold K 0-7 & 16-23; lanes16-31 hold K 8-15 & 24-31. N = lane&15.
DEV int b_koff(int idx, int hi) {
  int v = idx >> 1, e = idx & 1;
  return ((v & 3) * 2) + (hi ? 8 : 0) + ((v >> 2) * 16) + e;
}

DEV v16h loadA_f16(const _Float16* row, int k0, int hi) {
  v16h a;
#pragma unroll
  for (int i = 0; i < 16; ++i) a[i] = row[k0 + a_koff(i, hi)];
  return a;
}
DEV v16h loadB_f16(const _Float16* row, int k0, int hi) {
  v16h b;
#pragma unroll
  for (int i = 0; i < 16; ++i) b[i] = row[k0 + b_koff(i, hi)];
  return b;
}
DEV v8f wmma_f16(v16h a, v16h b, v8f c) {
  return __builtin_amdgcn_wmma_f32_16x16x32_f16(false, a, false, b, (short)0, c, false, false);
}

// =====================================================================
// K0: transpose weights to f16 [Dout][Din]; spatial_emb gather
// =====================================================================
__global__ void k_prep(const int* in_deg, const int* out_deg,
                       const float* in_emb, const float* out_emb,
                       const float* W, const float* Wq, const float* Wk,
                       const float* Wv, const float* Wo, const float* Wp,
                       _Float16* WT_W, _Float16* WT_Q, _Float16* WT_K,
                       _Float16* WT_V, _Float16* WT_O, _Float16* WT_P,
                       float* spatial) {
  int idx = blockIdx.x * blockDim.x + threadIdx.x;
  if (idx < 5 * 4096) {
    int mi = idx / 4096, i = idx % 4096;
    int dout = i >> 6, din = i & 63;
    const float* src = (mi == 0) ? W : (mi == 1) ? Wq : (mi == 2) ? Wk
                     : (mi == 3) ? Wv : Wo;
    _Float16* dst = (mi == 0) ? WT_W : (mi == 1) ? WT_Q : (mi == 2) ? WT_K
                  : (mi == 3) ? WT_V : WT_O;
    dst[i] = (_Float16)src[din * 64 + dout];
  } else if (idx < 5 * 4096 + 8192) {
    int i = idx - 20480;
    int dout = i >> 7, din = i & 127;
    WT_P[i] = (_Float16)Wp[din * 64 + dout];
  } else if (idx < 28672 + N * D) {
    int i = idx - 28672;
    int n = i >> 6, d = i & 63;
    spatial[i] = in_emb[in_deg[n] * 64 + d] + out_emb[out_deg[n] * 64 + d];
  }
}

// =====================================================================
// K1: per (sample, node-tile): Wh = x@W (store WhT f16, Wh1/Wh2),
//     q/k/v = (x+spatial)@W{q,k,v}+b (store q,k row-major/head, vT/head)
// =====================================================================
__global__ void __launch_bounds__(32)
k_proj(const float* __restrict__ inputs, const float* __restrict__ spatial,
       const _Float16* WT_W, const _Float16* WT_Q, const _Float16* WT_K,
       const _Float16* WT_V, const float* a1, const float* a2,
       const float* bq, const float* bk, const float* bv,
       float* Wh1, float* Wh2, _Float16* WhT,
       _Float16* qbuf, _Float16* kbuf, _Float16* vT) {
  const int s = blockIdx.x / NT, tile = blockIdx.x % NT;
  const int n0 = tile * 16;
  const int lane = threadIdx.x & 31;
  const int hi = lane >> 4;
  const int m = lane & 15;             // A row / C column index
  const int nodeA = min(n0 + m, N - 1);
  const float* xrow = inputs + ((size_t)s * N + nodeA) * D;
  const float* srow = spatial + (size_t)nodeA * D;

  v16h ax0, ax1, ay0, ay1;             // A frags for x and y=x+spatial
#pragma unroll
  for (int i = 0; i < 16; ++i) {
    int k = a_koff(i, hi);
    float x0 = xrow[k], x1 = xrow[32 + k];
    ax0[i] = (_Float16)x0;
    ax1[i] = (_Float16)x1;
    ay0[i] = (_Float16)(x0 + srow[k]);
    ay1[i] = (_Float16)(x1 + srow[32 + k]);
  }

  // ---- Wh = x @ W ----
  float h1[8], h2[8];
#pragma unroll
  for (int r = 0; r < 8; ++r) { h1[r] = 0.f; h2[r] = 0.f; }
#pragma unroll
  for (int ct = 0; ct < 4; ++ct) {
    const _Float16* wrow = WT_W + (size_t)(ct * 16 + m) * 64;
    v8f c = {};
    c = wmma_f16(ax0, loadB_f16(wrow, 0, hi), c);
    c = wmma_f16(ax1, loadB_f16(wrow, 32, hi), c);
    int d = ct * 16 + m;
    float a1d = a1[d], a2d = a2[d];
#pragma unroll
    for (int r = 0; r < 8; ++r) {
      float val = c[r];
      int node = n0 + (hi ? 8 : 0) + r;
      WhT[((size_t)s * 64 + d) * NP + node] = (_Float16)val;
      h1[r] += val * a1d;
      h2[r] += val * a2d;
    }
  }
  // reduce Wh·a1 / Wh·a2 across the 16 lanes of each half-wave
#pragma unroll
  for (int off = 1; off < 16; off <<= 1) {
#pragma unroll
    for (int r = 0; r < 8; ++r) {
      h1[r] += __shfl_xor(h1[r], off, 32);
      h2[r] += __shfl_xor(h2[r], off, 32);
    }
  }
  if (m == 0) {
#pragma unroll
    for (int r = 0; r < 8; ++r) {
      int node = n0 + (hi ? 8 : 0) + r;
      Wh1[s * NP + node] = h1[r];
      Wh2[s * NP + node] = h2[r];
    }
  }

  // ---- q, k, v projections (ct == head, dh == 16) ----
#pragma unroll
  for (int ct = 0; ct < 4; ++ct) {
    const _Float16* qw = WT_Q + (size_t)(ct * 16 + m) * 64;
    const _Float16* kw = WT_K + (size_t)(ct * 16 + m) * 64;
    const _Float16* vw = WT_V + (size_t)(ct * 16 + m) * 64;
    v8f cq = {}, ck = {}, cv = {};
    cq = wmma_f16(ay0, loadB_f16(qw, 0, hi), cq);
    cq = wmma_f16(ay1, loadB_f16(qw, 32, hi), cq);
    ck = wmma_f16(ay0, loadB_f16(kw, 0, hi), ck);
    ck = wmma_f16(ay1, loadB_f16(kw, 32, hi), ck);
    cv = wmma_f16(ay0, loadB_f16(vw, 0, hi), cv);
    cv = wmma_f16(ay1, loadB_f16(vw, 32, hi), cv);
    int d = ct * 16 + m;
    float bqd = bq[d], bkd = bk[d], bvd = bv[d];
#pragma unroll
    for (int r = 0; r < 8; ++r) {
      int node = n0 + (hi ? 8 : 0) + r;
      size_t hb = ((size_t)s * H + ct);
      qbuf[(hb * NP + node) * DH + m] = (_Float16)(cq[r] + bqd);
      kbuf[(hb * NP + node) * DH + m] = (_Float16)(ck[r] + bkd);
      vT[(hb * DH + m) * NP + node]   = (_Float16)(cv[r] + bvd);
    }
  }
}

// =====================================================================
// K2: GAT attention. e = leaky(Wh1_i + Wh2_j) masked by adj; softmax;
//     out1 = elu(attn @ Wh) -> f16 into cat[:, :, 0:64]
// =====================================================================
__global__ void __launch_bounds__(32)
k_gat(const float* __restrict__ Wh1g, const float* __restrict__ Wh2g,
      const int* __restrict__ adj, const _Float16* __restrict__ WhT,
      _Float16* cat) {
  const int s = blockIdx.x / NT, tile = blockIdx.x % NT;
  const int i0 = tile * 16;
  const int lane = threadIdx.x & 31;
  const int hi = lane >> 4;
  const int m = lane & 15;
  const int irow = i0 + m;
  const int irc = min(irow, N - 1);
  const float wh1 = Wh1g[s * NP + irow];
  const float* wh2p = Wh2g + s * NP;
  const int* adjrow = adj + (size_t)irc * N;

  // pass 1: online softmax stats, j split across half-waves
  float mx = -3.4e38f, sum = 0.f;
  const int jb = hi ? 358 : 0, je = hi ? N : 358;
  for (int j = jb; j < je; ++j) {
    float t = wh1 + wh2p[j];
    t = t > 0.f ? t : ALPHA * t;
    t = adjrow[j] ? t : NEGV;
    float nm = fmaxf(mx, t);
    sum = sum * __expf(mx - nm) + __expf(t - nm);
    mx = nm;
  }
  float mo = __shfl_xor(mx, 16, 32);
  float so = __shfl_xor(sum, 16, 32);
  float M = fmaxf(mx, mo);
  float S = sum * __expf(mx - M) + so * __expf(mo - M);
  float inv = 1.f / S;

  // pass 2: P tiles (16x32) x WhT tiles -> out1 (16x64)
  v8f c[4] = {};
  for (int jt = 0; jt < NP / 32; ++jt) {
    int j0 = jt * 32;
    v16h a;
#pragma unroll
    for (int i = 0; i < 16; ++i) {
      int j = j0 + a_koff(i, hi);
      float p = 0.f;
      if (j < N) {
        float t = wh1 + wh2p[j];
        t = t > 0.f ? t : ALPHA * t;
        t = adjrow[j] ? t : NEGV;
        p = __expf(t - M) * inv;
      }
      a[i] = (_Float16)p;
    }
#pragma unroll
    for (int ct = 0; ct < 4; ++ct) {
      const _Float16* wrow = WhT + ((size_t)s * 64 + ct * 16 + m) * NP + j0;
      c[ct] = wmma_f16(a, loadB_f16(wrow, 0, hi), c[ct]);
    }
  }
#pragma unroll
  for (int ct = 0; ct < 4; ++ct) {
    int d = ct * 16 + m;
#pragma unroll
    for (int r = 0; r < 8; ++r) {
      int node = i0 + (hi ? 8 : 0) + r;
      float v = c[ct][r];
      v = v > 0.f ? v : (__expf(v) - 1.f);   // ELU
      cat[((size_t)s * NP + node) * 128 + d] = (_Float16)v;
    }
  }
}

// =====================================================================
// K3: spatial MHA per (sample, tile, head): scores -> LDS -> softmax -> p@v
// =====================================================================
__global__ void __launch_bounds__(32)
k_mha(const _Float16* __restrict__ qbuf, const _Float16* __restrict__ kbuf,
      const _Float16* __restrict__ vTb, _Float16* obuf) {
  __shared__ float sS[16 * NP];        // 47104 B
  const int bid = blockIdx.x;
  const int s = bid / (NT * H);
  const int rem = bid % (NT * H);
  const int tile = rem / H, h = rem % H;
  const int i0 = tile * 16;
  const int lane = threadIdx.x & 31;
  const int hi = lane >> 4;
  const int m = lane & 15;
  const size_t hb = (size_t)s * H + h;

  // A frag: q rows (K=16 real, upper half zero), scaled by 1/sqrt(dh)=0.25
  const _Float16* qrow = qbuf + (hb * NP + (i0 + m)) * DH;
  v16h a;
#pragma unroll
  for (int i = 0; i < 16; ++i) {
    int k = a_koff(i, hi);
    a[i] = (k < DH) ? (_Float16)((float)qrow[k] * 0.25f) : (_Float16)0.f;
  }

  // scores tile row-band -> LDS
  for (int mt = 0; mt < NT; ++mt) {
    const _Float16* krow = kbuf + (hb * NP + (mt * 16 + m)) * DH;
    v16h b;
#pragma unroll
    for (int i = 0; i < 16; ++i) {
      int k = b_koff(i, hi);
      b[i] = (k < DH) ? krow[k] : (_Float16)0.f;
    }
    v8f c = {};
    c = wmma_f16(a, b, c);
#pragma unroll
    for (int r = 0; r < 8; ++r)
      sS[((hi ? 8 : 0) + r) * NP + mt * 16 + m] = c[r];
  }
  __syncthreads();

  // softmax stats over valid columns
  float mx = -3.4e38f, sum = 0.f;
  const int jb = hi ? 358 : 0, je = hi ? N : 358;
  for (int j = jb; j < je; ++j) {
    float t = sS[m * NP + j];
    float nm = fmaxf(mx, t);
    sum = sum * __expf(mx - nm) + __expf(t - nm);
    mx = nm;
  }
  float mo = __shfl_xor(mx, 16, 32);
  float so = __shfl_xor(sum, 16, 32);
  float M = fmaxf(mx, mo);
  float S = sum * __expf(mx - M) + so * __expf(mo - M);
  float inv = 1.f / S;

  // p @ v
  v8f c2 = {};
  for (int jt = 0; jt < NP / 32; ++jt) {
    int j0 = jt * 32;
    v16h a2;
#pragma unroll
    for (int i = 0; i < 16; ++i) {
      int j = j0 + a_koff(i, hi);
      float p = (j < N) ? __expf(sS[m * NP + j] - M) * inv : 0.f;
      a2[i] = (_Float16)p;
    }
    const _Float16* vrow = vTb + (hb * DH + m) * NP + j0;
    c2 = wmma_f16(a2, loadB_f16(vrow, 0, hi), c2);
  }
#pragma unroll
  for (int r = 0; r < 8; ++r) {
    int node = i0 + (hi ? 8 : 0) + r;
    obuf[((size_t)s * NP + node) * D + h * DH + m] = (_Float16)c2[r];
  }
}

// =====================================================================
// K4: out2 = o @ Wo + bo -> f16 into cat[:, :, 64:128]
// =====================================================================
__global__ void __launch_bounds__(32)
k_out2(const _Float16* __restrict__ obuf, const _Float16* WT_O,
       const float* bo, _Float16* cat) {
  const int s = blockIdx.x / NT, tile = blockIdx.x % NT;
  const int i0 = tile * 16;
  const int lane = threadIdx.x & 31;
  const int hi = lane >> 4;
  const int m = lane & 15;
  const _Float16* orow = obuf + ((size_t)s * NP + (i0 + m)) * D;
  v16h a0 = loadA_f16(orow, 0, hi), a1 = loadA_f16(orow, 32, hi);
#pragma unroll
  for (int ct = 0; ct < 4; ++ct) {
    const _Float16* w = WT_O + (size_t)(ct * 16 + m) * 64;
    v8f c = {};
    c = wmma_f16(a0, loadB_f16(w, 0, hi), c);
    c = wmma_f16(a1, loadB_f16(w, 32, hi), c);
    int d = ct * 16 + m;
    float b = bo[d];
#pragma unroll
    for (int r = 0; r < 8; ++r) {
      int node = i0 + (hi ? 8 : 0) + r;
      cat[((size_t)s * NP + node) * 128 + 64 + d] = (_Float16)(c[r] + b);
    }
  }
}

// =====================================================================
// K5: final = cat @ Wp + bp -> f32 output
// =====================================================================
__global__ void __launch_bounds__(32)
k_final(const _Float16* __restrict__ cat, const _Float16* WT_P,
        const float* bp, float* out) {
  const int s = blockIdx.x / NT, tile = blockIdx.x % NT;
  const int i0 = tile * 16;
  const int lane = threadIdx.x & 31;
  const int hi = lane >> 4;
  const int m = lane & 15;
  const _Float16* crow = cat + ((size_t)s * NP + (i0 + m)) * 128;
  v16h a[4];
#pragma unroll
  for (int kk = 0; kk < 4; ++kk) a[kk] = loadA_f16(crow, kk * 32, hi);
#pragma unroll
  for (int ct = 0; ct < 4; ++ct) {
    const _Float16* w = WT_P + (size_t)(ct * 16 + m) * 128;
    v8f c = {};
#pragma unroll
    for (int kk = 0; kk < 4; ++kk)
      c = wmma_f16(a[kk], loadB_f16(w, kk * 32, hi), c);
    int d = ct * 16 + m;
    float b = bp[d];
#pragma unroll
    for (int r = 0; r < 8; ++r) {
      int node = i0 + (hi ? 8 : 0) + r;
      if (node < N)
        out[((size_t)s * N + node) * D + d] = c[r] + b;
    }
  }
}

// =====================================================================
extern "C" void kernel_launch(void* const* d_in, const int* in_sizes, int n_in,
                              void* d_out, int out_size, void* d_ws, size_t ws_size,
                              hipStream_t stream) {
  (void)in_sizes; (void)n_in; (void)out_size; (void)ws_size;
  const float* inputs  = (const float*)d_in[0];
  const int*   adj     = (const int*)d_in[1];
  const int*   in_deg  = (const int*)d_in[2];
  const int*   out_deg = (const int*)d_in[3];
  const float* in_emb  = (const float*)d_in[4];
  const float* out_emb = (const float*)d_in[5];
  const float* W   = (const float*)d_in[6];
  const float* a1  = (const float*)d_in[7];
  const float* a2  = (const float*)d_in[8];
  const float* Wq  = (const float*)d_in[9];
  const float* bq  = (const float*)d_in[10];
  const float* Wk  = (const float*)d_in[11];
  const float* bk  = (const float*)d_in[12];
  const float* Wv  = (const float*)d_in[13];
  const float* bv  = (const float*)d_in[14];
  const float* Wo  = (const float*)d_in[15];
  const float* bo  = (const float*)d_in[16];
  const float* Wp  = (const float*)d_in[17];
  const float* bp  = (const float*)d_in[18];
  float* out = (float*)d_out;

  // carve workspace (256-B aligned chunks)
  char* p = (char*)d_ws;
  auto carve = [&](size_t bytes) -> char* {
    char* r = p;
    p += (bytes + 255) & ~(size_t)255;
    return r;
  };
  _Float16* WT_W = (_Float16*)carve(4096 * 2);
  _Float16* WT_Q = (_Float16*)carve(4096 * 2);
  _Float16* WT_K = (_Float16*)carve(4096 * 2);
  _Float16* WT_V = (_Float16*)carve(4096 * 2);
  _Float16* WT_O = (_Float16*)carve(4096 * 2);
  _Float16* WT_P = (_Float16*)carve(8192 * 2);
  float*    spatial = (float*)carve((size_t)N * D * 4);
  float*    Wh1  = (float*)carve((size_t)BS * NP * 4);
  float*    Wh2  = (float*)carve((size_t)BS * NP * 4);
  _Float16* WhT  = (_Float16*)carve((size_t)BS * D * NP * 2);
  _Float16* qbuf = (_Float16*)carve((size_t)BS * H * NP * DH * 2);
  _Float16* kbuf = (_Float16*)carve((size_t)BS * H * NP * DH * 2);
  _Float16* vT   = (_Float16*)carve((size_t)BS * H * DH * NP * 2);
  _Float16* obuf = (_Float16*)carve((size_t)BS * NP * D * 2);
  _Float16* cat  = (_Float16*)carve((size_t)BS * NP * 128 * 2);

  const int tasks0 = 5 * 4096 + 8192 + N * D;
  k_prep<<<(tasks0 + 255) / 256, 256, 0, stream>>>(
      in_deg, out_deg, in_emb, out_emb, W, Wq, Wk, Wv, Wo, Wp,
      WT_W, WT_Q, WT_K, WT_V, WT_O, WT_P, spatial);

  dim3 gTiles(BS * NT);
  k_proj<<<gTiles, 32, 0, stream>>>(inputs, spatial, WT_W, WT_Q, WT_K, WT_V,
                                    a1, a2, bq, bk, bv,
                                    Wh1, Wh2, WhT, qbuf, kbuf, vT);
  k_gat<<<gTiles, 32, 0, stream>>>(Wh1, Wh2, adj, WhT, cat);
  k_mha<<<dim3(BS * NT * H), 32, 0, stream>>>(qbuf, kbuf, vT, obuf);
  k_out2<<<gTiles, 32, 0, stream>>>(obuf, WT_O, bo, cat);
  k_final<<<gTiles, 32, 0, stream>>>(cat, WT_P, bp, out);
}
